// GeneralGNN_51170240365039
// MI455X (gfx1250) — compile-verified, matrix operand order checked by reading, not run on hardware
//
#include <hip/hip_runtime.h>
#include <hip/hip_bf16.h>
#include <math.h>

// ---------------- problem constants ----------------
#define N_NODES 10000
#define E_EDGES 320000
#define H_DIM   128
#define NV_DIM  16
#define HEADS   4
#define D_HEAD  32
#define NTILES  (E_EDGES / 16)   // 20000 edge tiles of 16

typedef _Float16 v16h __attribute__((ext_vector_type(16)));
typedef float    v8f  __attribute__((ext_vector_type(8)));
typedef unsigned int u32x4 __attribute__((ext_vector_type(4)));
typedef int i32x4 __attribute__((ext_vector_type(4)));
typedef int i32x8 __attribute__((ext_vector_type(8)));

// ---------------- workspace layout (bytes) ----------------
#define OFF_M      0                    // m_enc : N*8 u32      (320000 B)
#define OFF_S      320000               // s     : N*8 f32      (320000 B)
#define OFF_AGG    640000               // h_agg : N*128 f32    (5120000 B)
#define OFF_STATS  5760000              // dstats: 64 f32       (256 B)
#define OFF_LOGITS 5760256              // logits: E*8 f32      (10240000 B)
#define OFF_V      16000256             // V     : E*128 f16    (81920000 B)
#define OFF_W      97920256             // packed weights (f16, 462848 B)
#define OFF_HE16   98383104             // h_E f16 copy: E*256  (163840000 B)
#define OFF_HV16   262223104            // h_V f16 copy: N*128  (2560000 B)

// packed-weight sub-offsets, in _Float16 elements
#define PW_RED 0        // 320x256 -> 10*16*512
#define PW_B1  81920    // 384x128 -> 12*8*512
#define PW_B2  131072   // 128x128 -> 4*8*512
#define PW_B3  147456   // 128x16(pad) -> 4*1*512
#define PW_V1  149504   // 256x128 -> 8*8*512
#define PW_V2  182272
#define PW_V3  198656
#define PW_O   215040

// ---------------- helpers ----------------
__device__ __forceinline__ int rfl(int x) { return __builtin_amdgcn_readfirstlane(x); }

__device__ __forceinline__ unsigned fenc(float x) {
  unsigned b = __float_as_uint(x);
  return (b & 0x80000000u) ? ~b : (b | 0x80000000u);
}
__device__ __forceinline__ float fdec(unsigned e) {
  return __uint_as_float((e & 0x80000000u) ? (e & 0x7fffffffu) : ~e);
}

// ---- Tensor Data Mover: 2D tile (optionally gather-mode) global->LDS ----
// g0: count=1 (+ gather_mode bit31); lds_addr; global_addr[56:0]; type=2.
// g1: data_size=2B, pad_enable, pad_interval/amount; tensor/tile dims; stride0.
// g2/g3: zero (2D) or 16 x 16-bit gather row indices.
__device__ __forceinline__ void tdm_load_2d(unsigned g0w0, unsigned lds_addr,
                                            unsigned long long gaddr,
                                            unsigned w0flags,
                                            unsigned tensor_d0, unsigned tensor_d1,
                                            unsigned tile_d0, unsigned tile_d1,
                                            unsigned stride0,
                                            i32x4 g2, i32x4 g3) {
  u32x4 g0;
  g0[0] = (unsigned)rfl((int)g0w0);
  g0[1] = (unsigned)rfl((int)lds_addr);
  g0[2] = (unsigned)rfl((int)(unsigned)gaddr);
  g0[3] = (unsigned)rfl((int)(((unsigned)(gaddr >> 32) & 0x01FFFFFFu) | (2u << 30)));
  i32x8 g1;
  g1[0] = rfl((int)w0flags);
  g1[1] = rfl((int)((tensor_d0 & 0xFFFFu) << 16));
  g1[2] = rfl((int)((tensor_d0 >> 16) | ((tensor_d1 & 0xFFFFu) << 16)));
  g1[3] = rfl((int)((tensor_d1 >> 16) | (tile_d0 << 16)));
  g1[4] = rfl((int)tile_d1);           // tile_dim2 = 0
  g1[5] = rfl((int)stride0);           // tensor_dim0_stride[31:0]
  g1[6] = 0;                           // stride0 hi | stride1 lo (unused, 2D)
  g1[7] = 0;
#if defined(__clang_major__) && __clang_major__ >= 23
  i32x8 g4 = {};
  __builtin_amdgcn_tensor_load_to_lds(g0, g1, g2, g3, g4, 0);
#else
  __builtin_amdgcn_tensor_load_to_lds(g0, g1, g2, g3, 0);
#endif
}

// A fragment (16x32 f16) from an LDS row-major tile, ISA VGPR layout:
// lanes 0-15: row=lane, K = {kbase+0..7, kbase+16..23}
// lanes16-31: row=lane-16, K = {kbase+8..15, kbase+24..31}
__device__ __forceinline__ v16h a_frag_lds(const _Float16* base, int stride,
                                           int kbase, int lane) {
  const _Float16* p = base + (lane & 15) * stride + kbase + ((lane >> 4) << 3);
  v16h a;
#pragma unroll
  for (int j = 0; j < 8; ++j) { a[j] = p[j]; a[j + 8] = p[j + 16]; }
  return a;
}

// One 16x16 output tile: A(16xK f16 in LDS) x B(Kx16 f16 packed fragments)
__device__ __forceinline__ v8f gemm_tile(const _Float16* aBase, int astride,
                                         int kbase0, int KT,
                                         const _Float16* wpack, int nt, int lane) {
  v8f acc = {};
  for (int kt = 0; kt < KT; ++kt) {
    v16h a = a_frag_lds(aBase, astride, kbase0 + kt * 32, lane);
    v16h b = *((const v16h*)wpack + (size_t)(nt * KT + kt) * 32 + lane);
    acc = __builtin_amdgcn_wmma_f32_16x16x32_f16(
        /*neg_a=*/false, a, /*neg_b=*/false, b,
        /*c_mod=*/(short)0, acc, /*reuse_a=*/false, /*reuse_b=*/false);
  }
  return acc;
}

// Store a 16x16 f32 accumulator tile to LDS as f16 with bias + activation.
// ACT: 0 = none, 1 = relu, 2 = gelu(exact)
template <int ACT>
__device__ __forceinline__ void store_tile_lds(v8f acc, const float* bias, int nt,
                                               _Float16* dst, int dstride, int lane) {
  int col = nt * 16 + (lane & 15);
  int rbase = (lane >> 4) << 3;
  float b = bias[col];
#pragma unroll
  for (int r = 0; r < 8; ++r) {
    float x = acc[r] + b;
    if (ACT == 1) x = fmaxf(x, 0.0f);
    if (ACT == 2) x = 0.5f * x * (1.0f + erff(x * 0.70710678118654752f));
    dst[(rbase + r) * dstride + col] = (_Float16)x;
  }
}

// geometry: v_dst_t = frame * v_dst rows
__device__ __forceinline__ void compute_vdt(const float* __restrict__ F9,
                                            const float* __restrict__ vd,
                                            float vdt[16][3]) {
  float f[9];
#pragma unroll
  for (int i = 0; i < 9; ++i) f[i] = F9[i];
#pragma unroll
  for (int v = 0; v < 16; ++v) {
    float x = vd[v * 3 + 0], y = vd[v * 3 + 1], z = vd[v * 3 + 2];
#pragma unroll
    for (int i = 0; i < 3; ++i)
      vdt[v][i] = f[i * 3 + 0] * x + f[i * 3 + 1] * y + f[i * 3 + 2] * z;
  }
}

__device__ __forceinline__ void compute_vedge(int o, const float vdt[16][3],
                                              const float* __restrict__ vs,
                                              const float* __restrict__ Wv,
                                              float ve[3]) {
  ve[0] = vdt[o][0]; ve[1] = vdt[o][1]; ve[2] = vdt[o][2];
#pragma unroll
  for (int v = 0; v < 16; ++v) {
    float w = Wv[o * 32 + v];
    ve[0] += w * vdt[v][0]; ve[1] += w * vdt[v][1]; ve[2] += w * vdt[v][2];
  }
#pragma unroll
  for (int v = 0; v < 16; ++v) {
    float w = Wv[o * 32 + 16 + v];
    ve[0] += w * vs[v * 3 + 0]; ve[1] += w * vs[v * 3 + 1]; ve[2] += w * vs[v * 3 + 2];
  }
}

// ---------------- kernel: f32 -> f16 bulk convert (vectorized x4) ----------------
__global__ void __launch_bounds__(256)
cvt_f32_to_f16(const float* __restrict__ src, _Float16* __restrict__ dst, long n4) {
  long t = (long)blockIdx.x * 256 + threadIdx.x;
  if (t >= n4) return;
  const float4 q = *(const float4*)(src + t * 4);
  _Float16* p = dst + t * 4;
  p[0] = (_Float16)q.x; p[1] = (_Float16)q.y; p[2] = (_Float16)q.z; p[3] = (_Float16)q.w;
}

// ---------------- kernel: weight packing into B-fragment order ----------------
__global__ void __launch_bounds__(256)
pack_weights(const float* __restrict__ W, _Float16* __restrict__ out,
             int K, int Nsrc, int Npad) {
  int t = blockIdx.x * 256 + threadIdx.x;
  int KT = K >> 5, NT = Npad >> 4;
  if (t >= KT * NT * 32) return;
  int lane = t & 31;
  int tile = t >> 5;
  int n = ((tile / KT) * 16) + (lane & 15);
  int kb = (tile % KT) * 32 + ((lane >> 4) << 3);
  v16h vals;
#pragma unroll
  for (int j = 0; j < 8; ++j) {
    vals[j]     = (n < Nsrc) ? (_Float16)W[(size_t)(kb + j) * Nsrc + n] : (_Float16)0.0f;
    vals[j + 8] = (n < Nsrc) ? (_Float16)W[(size_t)(kb + 16 + j) * Nsrc + n] : (_Float16)0.0f;
  }
  *((v16h*)out + t) = vals;
}

// ---------------- kernel 1: dist statistics (sum, sumsq over E) ----------------
__global__ void __launch_bounds__(256)
k1_dist_stats(const int* __restrict__ eidx, const float* __restrict__ hvv,
              const float* __restrict__ frame, const float* __restrict__ W_vec,
              float* __restrict__ dstats) {
  __shared__ float sred[32];
  int t = threadIdx.x;
  if (t < 32) sred[t] = 0.0f;
  __syncthreads();
  int e = blockIdx.x * 256 + t;             // E divisible by 256
  int c = eidx[e];
  int d = eidx[E_EDGES + e];
  float vdt[16][3];
  compute_vdt(frame + (size_t)e * 9, hvv + (size_t)d * 48, vdt);
  const float* vs = hvv + (size_t)c * 48;
  for (int o = 0; o < 16; ++o) {
    float ve[3];
    compute_vedge(o, vdt, vs, W_vec, ve);
    float dist = sqrtf(ve[0] * ve[0] + ve[1] * ve[1] + ve[2] * ve[2]) + 1e-6f;
    atomicAdd(&sred[o], dist);              // ds_add_f32
    atomicAdd(&sred[16 + o], dist * dist);
  }
  __syncthreads();
  if (t < 32) atomicAdd(&dstats[t], sred[t]);
}

// ---------------- kernel 2: finalize batch-norm stats ----------------
__global__ void k2_finalize(float* __restrict__ dstats, const float* __restrict__ bn_g) {
  int o = threadIdx.x;
  if (o < 16) {
    float mu = dstats[o] * (1.0f / E_EDGES);
    float var = dstats[16 + o] * (1.0f / E_EDGES) - mu * mu;
    dstats[32 + o] = mu;
    dstats[48 + o] = rsqrtf(var + 1e-5f) * bn_g[o];   // fold gamma
  }
}

// ---------------- kernel 3: fused per-edge-tile GEMM chain ----------------
// 2 waves / block, each wave owns one tile of 16 edges.
// Per-wave LDS (halfs): X (16x384: [h_V | hE]), I (16x328: W_red input),
// buf0/buf1 (16x136 each) alias I once the W_red GEMM is done.
// h_E and h_V[center] tiles are delivered by the Tensor Data Mover:
//   - h_E: 2D tile 16x256 f16, LDS padding 36 DWORDs per 128-DWORD row -> I stride 328
//   - h_V: gather-mode, 16 x 16-bit row indices (centers), pad 128 DW -> X stride 384
#define WV_LDS 11392
__global__ void __launch_bounds__(64)
k3_edge_fused(const _Float16* __restrict__ hE16, const _Float16* __restrict__ hV16,
              const int* __restrict__ eidx, const float* __restrict__ hvv,
              const float* __restrict__ frame, const float* __restrict__ W_vec,
              const float* __restrict__ stats, const float* __restrict__ bn_b,
              const _Float16* __restrict__ wpack,
              const float* __restrict__ bb1, const float* __restrict__ bb2,
              const float* __restrict__ bb3, const float* __restrict__ bv1,
              const float* __restrict__ bv2, const float* __restrict__ bv3,
              float* __restrict__ logits, _Float16* __restrict__ Vout) {
  __shared__ _Float16 smem[2 * WV_LDS];
  const int lane = threadIdx.x & 31;
  const int wv = threadIdx.x >> 5;
  const int tile = blockIdx.x * 2 + wv;     // exactly NTILES waves in grid
  const int e0 = tile * 16;

  _Float16* X = smem + wv * WV_LDS;         // 16 x 384
  _Float16* I = X + 6144;                   // 16 x 328
  _Float16* buf0 = I;                       // 16 x 136 (aliases I, used after C)
  _Float16* buf1 = I + 2176;                // 16 x 136

  // ---- TDM: issue async tile loads first, overlap with geometry VALU ----
  {
    // h_E tile: 16 rows x 256 halfs into I cols [64,320), row stride 328 halfs.
    // pad_interval = 128 DWORDs (code 6); pad_amount = 36 DWORDs (code 35).
    unsigned flagsE = (1u << 16) | (1u << 20) | (6u << 22) | (35u << 25);
    i32x4 z = {};
    tdm_load_2d(/*g0w0=*/1u,
                (unsigned)(uintptr_t)(I + 64),
                (unsigned long long)(uintptr_t)(hE16 + (size_t)e0 * 256),
                flagsE, /*tensor_d0=*/256, /*tensor_d1=*/16,
                /*tile_d0=*/256, /*tile_d1=*/16, /*stride0=*/256, z, z);

    // h_V gather: 16 center rows x 128 halfs into X cols [0,128), stride 384.
    // pad_interval = 64 DWORDs (code 5); pad_amount = 128 DWORDs (code 127).
    i32x4 g2, g3;
#pragma unroll
    for (int j = 0; j < 4; ++j) {
      unsigned a0 = (unsigned)eidx[e0 + 2 * j];
      unsigned b0 = (unsigned)eidx[e0 + 2 * j + 1];
      g2[j] = rfl((int)(a0 | (b0 << 16)));
      unsigned a1 = (unsigned)eidx[e0 + 8 + 2 * j];
      unsigned b1 = (unsigned)eidx[e0 + 8 + 2 * j + 1];
      g3[j] = rfl((int)(a1 | (b1 << 16)));
    }
    unsigned flagsV = (1u << 16) | (1u << 20) | (5u << 22) | (127u << 25);
    tdm_load_2d(/*g0w0=*/1u | (1u << 31),          // gather_mode, 16-bit indices
                (unsigned)(uintptr_t)X,
                (unsigned long long)(uintptr_t)hV16,
                flagsV, /*tensor_d0=*/128, /*tensor_d1=*/(unsigned)N_NODES,
                /*tile_d0=*/128, /*tile_d1=*/16, /*stride0=*/128, g2, g3);
  }

  // ---- A: geometry -> I cols [0,64)  (2 lanes per edge, 8 outputs each)
  {
    const int el = lane >> 1;
    const int e = e0 + el;
    const int c = eidx[e];
    const int d = eidx[E_EDGES + e];
    float vdt[16][3];
    compute_vdt(frame + (size_t)e * 9, hvv + (size_t)d * 48, vdt);
    const float* vs = hvv + (size_t)c * 48;
    const int obase = (lane & 1) * 8;
    for (int oo = 0; oo < 8; ++oo) {
      int o = obase + oo;
      float ve[3];
      compute_vedge(o, vdt, vs, W_vec, ve);
      float dist = sqrtf(ve[0] * ve[0] + ve[1] * ve[1] + ve[2] * ve[2]) + 1e-6f;
      float inv = 1.0f / dist;
      I[el * 328 + o * 3 + 0] = (_Float16)(ve[0] * inv);
      I[el * 328 + o * 3 + 1] = (_Float16)(ve[1] * inv);
      I[el * 328 + o * 3 + 2] = (_Float16)(ve[2] * inv);
      float dbn = (dist - stats[32 + o]) * stats[48 + o] + bn_b[o];
      I[el * 328 + 48 + o] = (_Float16)dbn;
    }
  }

  if (lane == 0 && tile + 1 < NTILES)
    __builtin_prefetch(hE16 + (size_t)(e0 + 16) * 256, 0, 0);  // global_prefetch_b8

  // DMA tiles must have landed before the GEMMs read them.
  __builtin_amdgcn_s_wait_tensorcnt(0);

  // ---- C: hE = I(16x320) @ W_red -> X cols [128,384) (f16, no bias)
  for (int nt = 0; nt < 16; ++nt) {
    v8f acc = gemm_tile(I, 328, 0, 10, wpack + PW_RED, nt, lane);
    int col = nt * 16 + (lane & 15);
    int rbase = (lane >> 4) << 3;
#pragma unroll
    for (int r = 0; r < 8; ++r)
      X[(rbase + r) * 384 + 128 + col] = (_Float16)acc[r];
  }

  // ---- D: x1 = relu(X(16x384) @ Wb1 + bb1) -> buf0
  for (int nt = 0; nt < 8; ++nt) {
    v8f acc = gemm_tile(X, 384, 0, 12, wpack + PW_B1, nt, lane);
    store_tile_lds<1>(acc, bb1, nt, buf0, 136, lane);
  }
  // ---- E: x2 = relu(x1 @ Wb2 + bb2) -> buf1
  for (int nt = 0; nt < 8; ++nt) {
    v8f acc = gemm_tile(buf0, 136, 0, 4, wpack + PW_B2, nt, lane);
    store_tile_lds<1>(acc, bb2, nt, buf1, 136, lane);
  }
  // ---- F: logits = (x2 @ Wb3 + bb3) / sqrt(D) -> global (8 valid cols)
  {
    v8f acc = gemm_tile(buf1, 136, 0, 4, wpack + PW_B3, 0, lane);
    int n = lane & 15;
    int rbase = (lane >> 4) << 3;
    if (n < 8) {
      float b = bb3[n];
#pragma unroll
      for (int r = 0; r < 8; ++r)
        logits[(size_t)(e0 + rbase + r) * 8 + n] = (acc[r] + b) * 0.17677669529663687f;
    }
  }
  // ---- G: v1 = gelu(hE(16x256) @ Wv1 + bv1) -> buf0 (reads X cols 128..383)
  for (int nt = 0; nt < 8; ++nt) {
    v8f acc = gemm_tile(X, 384, 128, 8, wpack + PW_V1, nt, lane);
    store_tile_lds<2>(acc, bv1, nt, buf0, 136, lane);
  }
  // ---- H: v2 = gelu(v1 @ Wv2 + bv2) -> buf1
  for (int nt = 0; nt < 8; ++nt) {
    v8f acc = gemm_tile(buf0, 136, 0, 4, wpack + PW_V2, nt, lane);
    store_tile_lds<2>(acc, bv2, nt, buf1, 136, lane);
  }
  // ---- I: V = v2 @ Wv3 + bv3 -> global f16
  for (int nt = 0; nt < 8; ++nt) {
    v8f acc = gemm_tile(buf1, 136, 0, 4, wpack + PW_V3, nt, lane);
    int col = nt * 16 + (lane & 15);
    int rbase = (lane >> 4) << 3;
    float b = bv3[col];
#pragma unroll
    for (int r = 0; r < 8; ++r)
      Vout[(size_t)(e0 + rbase + r) * 128 + col] = (_Float16)(acc[r] + b);
  }
}

// ---------------- kernel 4: segment max (monotone-uint atomicMax) ----------------
__global__ void __launch_bounds__(256)
k4_segmax(const float* __restrict__ logits, const int* __restrict__ eidx,
          unsigned* __restrict__ m_enc) {
  int t = blockIdx.x * 256 + threadIdx.x;   // t = e*8 + h, E*8 divisible by 256
  int e = t >> 3, h = t & 7;
  int c = eidx[e];
  atomicMax(&m_enc[c * 8 + h], fenc(logits[t]));
}

// ---------------- kernel 5: segment sum of exp(logit - m) ----------------
__global__ void __launch_bounds__(256)
k5_segsum(const float* __restrict__ logits, const int* __restrict__ eidx,
          const unsigned* __restrict__ m_enc, float* __restrict__ s) {
  int t = blockIdx.x * 256 + threadIdx.x;
  int e = t >> 3, h = t & 7;
  int c = eidx[e];
  float m = fdec(m_enc[c * 8 + h]);
  atomicAdd(&s[c * 8 + h], expf(logits[t] - m));
}

// ---------------- kernel 6: weighted aggregation into h_agg ----------------
__global__ void __launch_bounds__(256)
k6_aggregate(const float* __restrict__ logits, const int* __restrict__ eidx,
             const unsigned* __restrict__ m_enc, const float* __restrict__ s,
             const _Float16* __restrict__ V, float* __restrict__ h_agg) {
  size_t t = (size_t)blockIdx.x * 256 + threadIdx.x;   // t = e*128 + i
  int e = (int)(t >> 7);
  int i = (int)(t & 127);
  int h = i >> 5;                                      // first 4 of 8 softmax cols
  int c = eidx[e];
  float m = fdec(m_enc[c * 8 + h]);
  float att = expf(logits[(size_t)e * 8 + h] - m) / s[c * 8 + h];
  atomicAdd(&h_agg[(size_t)c * 128 + i], att * (float)V[t]);
}

// ---------------- kernel 7: out = h_agg @ W_O (WMMA) ----------------
__global__ void __launch_bounds__(64)
k7_out(const float* __restrict__ h_agg, const _Float16* __restrict__ wo_pack,
       float* __restrict__ out) {
  __shared__ _Float16 smem[2 * 16 * 136];
  const int lane = threadIdx.x & 31;
  const int wv = threadIdx.x >> 5;
  const int tile = blockIdx.x * 2 + wv;
  if (tile >= N_NODES / 16) return;        // per-wave LDS only; no barriers
  _Float16* A = smem + wv * (16 * 136);
  const int r0 = tile * 16;
  for (int idx = lane; idx < 512; idx += 32) {
    int row = idx >> 5, c4 = (idx & 31) << 2;
    const float4 q = *(const float4*)(h_agg + (size_t)(r0 + row) * 128 + c4);
    _Float16* p = A + row * 136 + c4;
    p[0] = (_Float16)q.x; p[1] = (_Float16)q.y; p[2] = (_Float16)q.z; p[3] = (_Float16)q.w;
  }
  for (int nt = 0; nt < 8; ++nt) {
    v8f acc = gemm_tile(A, 136, 0, 4, wo_pack, nt, lane);
    int col = nt * 16 + (lane & 15);
    int rbase = (lane >> 4) << 3;
#pragma unroll
    for (int r = 0; r < 8; ++r)
      out[(size_t)(r0 + rbase + r) * 128 + col] = acc[r];
  }
}

// ---------------- launch ----------------
extern "C" void kernel_launch(void* const* d_in, const int* in_sizes, int n_in,
                              void* d_out, int out_size, void* d_ws, size_t ws_size,
                              hipStream_t stream) {
  const float* h_V   = (const float*)d_in[0];
  const float* h_E   = (const float*)d_in[1];
  const int*   eidx  = (const int*)d_in[2];
  const float* hvv   = (const float*)d_in[3];
  const float* frame = (const float*)d_in[4];
  const float* W_vec = (const float*)d_in[5];
  const float* W_red = (const float*)d_in[6];
  const float* Wb1   = (const float*)d_in[7];
  const float* bb1   = (const float*)d_in[8];
  const float* Wb2   = (const float*)d_in[9];
  const float* bb2   = (const float*)d_in[10];
  const float* Wb3   = (const float*)d_in[11];
  const float* bb3   = (const float*)d_in[12];
  const float* Wv1   = (const float*)d_in[13];
  const float* bv1   = (const float*)d_in[14];
  const float* Wv2   = (const float*)d_in[15];
  const float* bv2   = (const float*)d_in[16];
  const float* Wv3   = (const float*)d_in[17];
  const float* bv3   = (const float*)d_in[18];
  const float* W_O   = (const float*)d_in[19];
  const float* bn_g  = (const float*)d_in[20];
  const float* bn_b  = (const float*)d_in[21];

  char* ws = (char*)d_ws;
  unsigned* m_enc = (unsigned*)(ws + OFF_M);
  float*    s_sum = (float*)(ws + OFF_S);
  float*    h_agg = (float*)(ws + OFF_AGG);
  float*    dstat = (float*)(ws + OFF_STATS);
  float*    lgits = (float*)(ws + OFF_LOGITS);
  _Float16* Vbuf  = (_Float16*)(ws + OFF_V);
  _Float16* wpack = (_Float16*)(ws + OFF_W);
  _Float16* hE16  = (_Float16*)(ws + OFF_HE16);
  _Float16* hV16  = (_Float16*)(ws + OFF_HV16);

  // zero: m_enc (encoded floor), s, h_agg, dstats
  hipMemsetAsync(ws, 0, OFF_STATS + 256, stream);

  // bulk f32 -> f16 copies feeding the TDM pipeline
  cvt_f32_to_f16<<<(E_EDGES * 256 / 4) / 256, 256, 0, stream>>>(h_E, hE16,
                                                               (long)E_EDGES * 64);
  cvt_f32_to_f16<<<(N_NODES * 128 / 4) / 256, 256, 0, stream>>>(h_V, hV16,
                                                               (long)N_NODES * 32);

  // pack weights f32 -> f16 B-fragment order (tiny; L2-resident afterwards)
  struct Pk { const float* w; _Float16* o; int K, Nsrc, Npad; } pks[8] = {
      {W_red, wpack + PW_RED, 320, 256, 256}, {Wb1, wpack + PW_B1, 384, 128, 128},
      {Wb2, wpack + PW_B2, 128, 128, 128},    {Wb3, wpack + PW_B3, 128, 8, 16},
      {Wv1, wpack + PW_V1, 256, 128, 128},    {Wv2, wpack + PW_V2, 128, 128, 128},
      {Wv3, wpack + PW_V3, 128, 128, 128},    {W_O, wpack + PW_O, 128, 128, 128}};
  for (int i = 0; i < 8; ++i) {
    int total = (pks[i].K >> 5) * (pks[i].Npad >> 4) * 32;
    pack_weights<<<(total + 255) / 256, 256, 0, stream>>>(
        pks[i].w, pks[i].o, pks[i].K, pks[i].Nsrc, pks[i].Npad);
  }

  k1_dist_stats<<<E_EDGES / 256, 256, 0, stream>>>(eidx, hvv, frame, W_vec, dstat);
  k2_finalize<<<1, 32, 0, stream>>>(dstat, bn_g);

  k3_edge_fused<<<NTILES / 2, 64, 0, stream>>>(
      hE16, hV16, eidx, hvv, frame, W_vec, dstat, bn_b, wpack,
      bb1, bb2, bb3, bv1, bv2, bv3, lgits, Vbuf);

  k4_segmax<<<(E_EDGES * 8) / 256, 256, 0, stream>>>(lgits, eidx, m_enc);
  k5_segsum<<<(E_EDGES * 8) / 256, 256, 0, stream>>>(lgits, eidx, m_enc, s_sum);
  k6_aggregate<<<(E_EDGES / 2), 256, 0, stream>>>(lgits, eidx, m_enc, s_sum, Vbuf, h_agg);

  k7_out<<<(N_NODES / 16 + 1) / 2, 64, 0, stream>>>(h_agg, wpack + PW_O, (float*)d_out);
}
